// SelfAttention_65481071398389
// MI455X (gfx1250) — matrix operand order repaired and checked by reading, not compiled, not run on hardware
//
#include <hip/hip_runtime.h>
#include <hip/hip_bf16.h>
#include <math.h>

// Problem constants (match reference)
#define EMBED  1024
#define NHEAD  16
#define HDIM   64
#define BATCH  2
#define SEQ    2048
#define M_TOT  (BATCH * SEQ)   // 4096
#define N_TOT  (3 * EMBED)     // 3072
#define K_TOT  EMBED           // 1024

typedef __attribute__((ext_vector_type(16))) __bf16 v16bf;
typedef __attribute__((ext_vector_type(8)))  __bf16 v8bf;
typedef __attribute__((ext_vector_type(8)))  float  v8f;
typedef __attribute__((ext_vector_type(4)))  float  v4f;

union FragBF {
    v16bf  v;
    v8bf   h[2];
    __bf16 e[16];
};

__device__ __forceinline__ v8f wmma_bf16(v16bf a, v16bf b, v8f c) {
    // D(f32 16x16) = A(bf16 16x32) * B(bf16 32x16) + C
    return __builtin_amdgcn_wmma_f32_16x16x32_bf16(
        false, a, false, b, (short)0, c, false, false);
}

// ---------------------------------------------------------------------------
// Kernel 0: bulk fp32 -> bf16 convert (8 elements / thread, b128 in, b128 out)
// ---------------------------------------------------------------------------
__global__ __launch_bounds__(256) void cvt_bf16(const float* __restrict__ src,
                                                __bf16* __restrict__ dst, int n) {
    const int i = (blockIdx.x * 256 + threadIdx.x) * 8;
    if (i + 8 > n) return;
    const v4f f0 = *(const v4f*)(src + i);
    const v4f f1 = *(const v4f*)(src + i + 4);
    v8bf o;
#pragma unroll
    for (int k = 0; k < 4; ++k) { o[k] = (__bf16)f0[k]; o[4 + k] = (__bf16)f1[k]; }
    *(v8bf*)(dst + i) = o;
}

// ---------------------------------------------------------------------------
// Kernel 1: fused QKV projection on bf16 operands.
//   qkv[m][n] = sum_k X[m][k] * W[n][k] + bias[n]
// Block tile 128(M) x 128(N); wave tile 32(M) x 64(N) = 2 A x 4 B fragments
// -> 8 WMMAs per K-step of 32, all fragment loads straight b128 from bf16.
// Epilogue scatters bf16 Q,K as [H,B,S,D] and V transposed as [H,B,D,S].
// ---------------------------------------------------------------------------
__global__ __launch_bounds__(256) void qkv_gemm(
    const __bf16* __restrict__ Xb,   // [4096][1024] bf16
    const __bf16* __restrict__ Wb,   // [3072][1024] bf16
    const float*  __restrict__ bias, // [3072] fp32
    __bf16* __restrict__ Qb,         // [H][B][S][D]
    __bf16* __restrict__ Kb,         // [H][B][S][D]
    __bf16* __restrict__ Vt)         // [H][B][D][S]
{
    const int wave = threadIdx.x >> 5;
    const int lane = threadIdx.x & 31;
    const int m16  = lane & 15;
    const int half = lane >> 4;
    const int wm   = wave & 3;                     // 4 waves along M
    const int wn   = wave >> 2;                    // 2 waves along N
    const int mbase = blockIdx.x * 128 + wm * 32;  // 32 blocks in M
    const int nbase = blockIdx.y * 128 + wn * 64;  // 24 blocks in N

    v8f acc[2][4] = {};
    const __bf16* arow0 = Xb + (size_t)(mbase + m16) * K_TOT;
    const __bf16* arow1 = Xb + (size_t)(mbase + 16 + m16) * K_TOT;

    for (int k = 0; k < K_TOT; k += 32) {
        if (k + 32 < K_TOT)
            __builtin_prefetch(arow0 + k + 32, 0, 0);   // global_prefetch_b8
        // A fragments (16 contiguous K per lane-half)
        FragBF a0, a1;
        a0.h[0] = *(const v8bf*)(arow0 + k + half * 16);
        a0.h[1] = *(const v8bf*)(arow0 + k + half * 16 + 8);
        a1.h[0] = *(const v8bf*)(arow1 + k + half * 16);
        a1.h[1] = *(const v8bf*)(arow1 + k + half * 16 + 8);
#pragma unroll
        for (int dt = 0; dt < 4; ++dt) {
            // B fragment: column n, 16 contiguous K per lane-half (32B aligned)
            const v16bf bm = *(const v16bf*)(
                Wb + (size_t)(nbase + dt * 16 + m16) * K_TOT + k + half * 16);
            acc[0][dt] = wmma_bf16(a0.v, bm, acc[0][dt]);
            acc[1][dt] = wmma_bf16(a1.v, bm, acc[1][dt]);
        }
    }

    // Epilogue: bias add, bf16 convert, scatter into Q / K / V^T layouts.
#pragma unroll
    for (int mt = 0; mt < 2; ++mt) {
#pragma unroll
        for (int dt = 0; dt < 4; ++dt) {
#pragma unroll
            for (int v = 0; v < 8; ++v) {
                const int gm = mbase + mt * 16 + v + 8 * half; // row in [0,4096)
                const int n  = nbase + dt * 16 + m16;          // col in [0,3072)
                const float val = acc[mt][dt][v] + bias[n];
                const int b_ = gm >> 11, s_ = gm & (SEQ - 1);  // m = b*S + s
                const int which = n >> 10;                     // 0=Q 1=K 2=V
                const int rem = n & (EMBED - 1);
                const int h = rem >> 6, d = rem & (HDIM - 1);
                const __bf16 bv = (__bf16)val;
                const size_t hbi = (size_t)h * BATCH + b_;
                if (which == 0)
                    Qb[(hbi * SEQ + s_) * HDIM + d] = bv;
                else if (which == 1)
                    Kb[(hbi * SEQ + s_) * HDIM + d] = bv;
                else
                    Vt[(hbi * HDIM + d) * SEQ + s_] = bv;
            }
        }
    }
}

// ---------------------------------------------------------------------------
// Kernel 2: flash attention (online softmax), K/V tiles staged through LDS
// and shared by all 8 waves of the block. One block per (128 q-rows, h*B+b);
// each wave owns 16 q-rows x D=64. Per 32-key step: 4 WMMAs (Q K^T) +
// softmax update + 4 WMMAs (P V). P round-trips a per-wave LDS slab to turn
// the C-fragment layout into the A-fragment layout.
// LDS row paddings chosen so 16-lane b128 groups hit distinct banks:
//   Ks stride 88 (44*r mod 64 distinct), Vs/pl stride 40 (20*r mod 64 distinct),
// all strides 16B multiples so ds b128 stays aligned.
// ---------------------------------------------------------------------------
__global__ __launch_bounds__(256) void flash_attn(
    const __bf16* __restrict__ Qb,   // [H][B][S][D]
    const __bf16* __restrict__ Kb,   // [H][B][S][D]
    const __bf16* __restrict__ Vt,   // [H][B][D][S]
    float* __restrict__ out)         // flat [S][D][H][B] == [B][S][E]
{
    __shared__ __align__(16) __bf16 Ks[32][88];     // keys x dims   (5.5 KB)
    __shared__ __align__(16) __bf16 Vs[64][40];     // dims x keys   (5 KB)
    __shared__ __align__(16) __bf16 pl[8][16][40];  // per-wave P    (10 KB)

    const int tid  = threadIdx.x;
    const int wave = tid >> 5;
    const int lane = tid & 31;
    const int m16  = lane & 15;
    const int half = lane >> 4;
    const int hb   = blockIdx.y;                    // h*B + b, 0..31
    const int h    = hb >> 1, b = hb & 1;
    const int qbase = blockIdx.x * 128 + wave * 16;

    const __bf16* Qp = Qb + (size_t)hb * SEQ * HDIM;
    const __bf16* Kp = Kb + (size_t)hb * SEQ * HDIM;
    const __bf16* Vp = Vt + (size_t)hb * HDIM * SEQ;

    // Q A-fragments for dims 0..31 (qa0) and 32..63 (qa1)
    FragBF qa0, qa1;
    {
        const __bf16* r = Qp + (size_t)(qbase + m16) * HDIM;
        qa0.h[0] = *(const v8bf*)(r + half * 16);
        qa0.h[1] = *(const v8bf*)(r + half * 16 + 8);
        qa1.h[0] = *(const v8bf*)(r + 32 + half * 16);
        qa1.h[1] = *(const v8bf*)(r + 32 + half * 16 + 8);
    }

    float rmax[8], rsum[8];
    v8f o[4] = {};
#pragma unroll
    for (int v = 0; v < 8; ++v) { rmax[v] = -3.0e38f; rsum[v] = 0.f; }
    const float scale = 0.125f;  // 1/sqrt(64)

    // Cooperative staging indices (256 threads, 8 bf16 each per tile)
    const int krow = tid >> 3, kcol = (tid & 7) * 8;   // Ks: 32 x 64
    const int vrow = tid >> 2, vcol = (tid & 3) * 8;   // Vs: 64 x 32

    for (int j = 0; j < SEQ; j += 32) {
        __syncthreads();   // previous tile fully consumed
        *(v8bf*)&Ks[krow][kcol] = *(const v8bf*)(Kp + (size_t)(j + krow) * HDIM + kcol);
        *(v8bf*)&Vs[vrow][vcol] = *(const v8bf*)(Vp + (size_t)vrow * SEQ + j + vcol);
        __syncthreads();

        // --- scores: C(16x32) = Q(16x64) * K^T(64x32), two 16x16 C tiles ---
        FragBF kb00, kb01, kb10, kb11;    // B layout: col=key, 16 contiguous dims/half
        kb00.h[0] = *(const v8bf*)&Ks[m16][half * 16];
        kb00.h[1] = *(const v8bf*)&Ks[m16][half * 16 + 8];
        kb01.h[0] = *(const v8bf*)&Ks[m16][32 + half * 16];
        kb01.h[1] = *(const v8bf*)&Ks[m16][32 + half * 16 + 8];
        kb10.h[0] = *(const v8bf*)&Ks[16 + m16][half * 16];
        kb10.h[1] = *(const v8bf*)&Ks[16 + m16][half * 16 + 8];
        kb11.h[0] = *(const v8bf*)&Ks[16 + m16][32 + half * 16];
        kb11.h[1] = *(const v8bf*)&Ks[16 + m16][32 + half * 16 + 8];
        v8f c0 = {}, c1 = {};
        c0 = wmma_bf16(qa0.v, kb00.v, c0);
        c0 = wmma_bf16(qa1.v, kb01.v, c0);
        c1 = wmma_bf16(qa0.v, kb10.v, c1);
        c1 = wmma_bf16(qa1.v, kb11.v, c1);

        // --- online softmax over the 32 new columns ---
#pragma unroll
        for (int v = 0; v < 8; ++v) {
            const float s0 = c0[v] * scale, s1 = c1[v] * scale;
            float t = fmaxf(s0, s1);
            t = fmaxf(t, __shfl_xor(t, 1, 32));
            t = fmaxf(t, __shfl_xor(t, 2, 32));
            t = fmaxf(t, __shfl_xor(t, 4, 32));
            t = fmaxf(t, __shfl_xor(t, 8, 32));   // row max within lane-half
            const float mnew  = fmaxf(rmax[v], t);
            const float alpha = __expf(rmax[v] - mnew);
            rmax[v] = mnew;
            const float p0 = __expf(s0 - mnew);
            const float p1 = __expf(s1 - mnew);
            float ps = p0 + p1;
            ps += __shfl_xor(ps, 1, 32);
            ps += __shfl_xor(ps, 2, 32);
            ps += __shfl_xor(ps, 4, 32);
            ps += __shfl_xor(ps, 8, 32);
            rsum[v] = rsum[v] * alpha + ps;
            o[0][v] *= alpha; o[1][v] *= alpha;
            o[2][v] *= alpha; o[3][v] *= alpha;
            const int row = v + 8 * half;
            pl[wave][row][m16]      = (__bf16)p0;   // ds_store_b16
            pl[wave][row][m16 + 16] = (__bf16)p1;
        }
        asm volatile("s_wait_dscnt 0" ::: "memory");

        // Re-read P tile in A-fragment layout (16 rows x 32 keys)
        FragBF pa;
        pa.h[0] = *(const v8bf*)&pl[wave][m16][half * 16];
        pa.h[1] = *(const v8bf*)&pl[wave][m16][half * 16 + 8];

        // --- O += P(16x32) * V(32x64): 4 N-tiles from the shared V^T tile ---
#pragma unroll
        for (int dt = 0; dt < 4; ++dt) {
            FragBF vb;
            vb.h[0] = *(const v8bf*)&Vs[dt * 16 + m16][half * 16];
            vb.h[1] = *(const v8bf*)&Vs[dt * 16 + m16][half * 16 + 8];
            o[dt] = wmma_bf16(pa.v, vb.v, o[dt]);
        }
    }

    // Normalize and write with the reference's [S,D,H,B]-flat permutation.
#pragma unroll
    for (int v = 0; v < 8; ++v) {
        const float inv = 1.0f / rsum[v];
        const int s_ = qbase + v + 8 * half;
#pragma unroll
        for (int dt = 0; dt < 4; ++dt) {
            const int d = dt * 16 + m16;
            out[(size_t)s_ * (HDIM * NHEAD * BATCH)
                + (size_t)d * (NHEAD * BATCH)
                + (size_t)h * BATCH + b] = o[dt][v] * inv;
        }
    }
}

// ---------------------------------------------------------------------------
extern "C" void kernel_launch(void* const* d_in, const int* in_sizes, int n_in,
                              void* d_out, int out_size, void* d_ws, size_t ws_size,
                              hipStream_t stream) {
    (void)in_sizes; (void)n_in; (void)out_size; (void)ws_size;
    const float* W    = (const float*)d_in[0];  // rotation_params [3072,1024]
    const float* bias = (const float*)d_in[1];  // entangle_params [3072]
    const float* X    = (const float*)d_in[2];  // inputs [2,2048,1024]
    float* out = (float*)d_out;

    // Workspace layout (bf16): Q | K | V^T | Xb | Wb  (~40 MB total)
    const size_t qkv_e = (size_t)NHEAD * BATCH * SEQ * HDIM;  // 4,194,304
    const size_t x_e   = (size_t)M_TOT * K_TOT;               // 4,194,304
    const size_t w_e   = (size_t)N_TOT * K_TOT;               // 3,145,728
    __bf16* Qb = (__bf16*)d_ws;
    __bf16* Kb = Qb + qkv_e;
    __bf16* Vt = Kb + qkv_e;
    __bf16* Xb = Vt + qkv_e;
    __bf16* Wb = Xb + x_e;

    cvt_bf16<<<(int)(x_e / (8 * 256)), 256, 0, stream>>>(X, Xb, (int)x_e);
    cvt_bf16<<<(int)(w_e / (8 * 256)), 256, 0, stream>>>(W, Wb, (int)w_e);
    qkv_gemm<<<dim3(M_TOT / 128, N_TOT / 128), 256, 0, stream>>>(Xb, Wb, bias,
                                                                 Qb, Kb, Vt);
    flash_attn<<<dim3(SEQ / 128, NHEAD * BATCH), 256, 0, stream>>>(Qb, Kb, Vt, out);
}
